// Net_25752623907118
// MI455X (gfx1250) — matrix operand order repaired and checked by reading, not compiled, not run on hardware
//
#include <hip/hip_runtime.h>

typedef __attribute__((ext_vector_type(16))) _Float16 v16h;
typedef __attribute__((ext_vector_type(8)))  _Float16 v8h;
typedef __attribute__((ext_vector_type(8)))  float    v8f;
typedef __attribute__((ext_vector_type(4)))  float    v4f;

// ---------------------------------------------------------------------------
// Small elementwise / prep kernels
// ---------------------------------------------------------------------------
__global__ void k_fill_f32(float* __restrict__ p, float v, int n) {
    int i = blockIdx.x * blockDim.x + threadIdx.x;
    if (i < n) p[i] = v;
}

__global__ void k_deg_accum(float* __restrict__ deg, const int* __restrict__ dst, int E) {
    int e = blockIdx.x * blockDim.x + threadIdx.x;
    if (e < E) atomicAdd(&deg[dst[e]], 1.0f);
}

__global__ void k_rsqrt_inplace(float* __restrict__ p, int n) {
    int i = blockIdx.x * blockDim.x + threadIdx.x;
    if (i < n) p[i] = rsqrtf(p[i]);
}

__global__ void k_cvt_f32_to_f16(_Float16* __restrict__ o, const float* __restrict__ in, int n) {
    int i = blockIdx.x * blockDim.x + threadIdx.x;
    if (i < n) o[i] = (_Float16)in[i];
}

// W is [K][Ncol] row-major f32; produce Wt [Ncol][K] f16 (contiguous K per column)
__global__ void k_wtrans_f16(_Float16* __restrict__ Wt, const float* __restrict__ W,
                             int K, int Ncol) {
    int i = blockIdx.x * blockDim.x + threadIdx.x;
    if (i >= K * Ncol) return;
    int k = i / Ncol, n = i % Ncol;
    Wt[(size_t)n * K + k] = (_Float16)W[i];
}

__global__ void k_relu_cvt_f16(_Float16* __restrict__ o, const float* __restrict__ in, int n) {
    int i = blockIdx.x * blockDim.x + threadIdx.x;
    if (i < n) {
        float v = in[i];
        o[i] = (_Float16)(v > 0.0f ? v : 0.0f);
    }
}

// ---------------------------------------------------------------------------
// WMMA GEMM: C[M x Ncol] = A[M x K] * Bt^T, A f16 row-major, Bt f16 [Ncol][K].
// One wave per 16x16 C tile; waves-in-block cover the Ncol tiles.
// VGPR layouts per CDNA5 ISA 7.12.2 (16-bit A 16x32; B 32x16; f32 C 16x16).
// (Round-1 codegen confirmed: fully unrolled, 8x v_wmma_f32_16x16x32_f16 with
//  clause'd b128 loads pipelined against WMMA issue — keep unchanged.)
// ---------------------------------------------------------------------------
__global__ void k_gemm_f16_wmma(const _Float16* __restrict__ A,
                                const _Float16* __restrict__ Bt,
                                float* __restrict__ C,
                                int M, int K, int Ncol) {
    const int lane = threadIdx.x & 31;
    const int wave = threadIdx.x >> 5;
    const int m0 = blockIdx.x * 16;
    const int n0 = wave * 16;

    int rowA = m0 + (lane & 15);
    if (rowA >= M) rowA = M - 1;                 // clamp (M=50000 is exact anyway)
    const int colB = n0 + (lane & 15);
    const int kA = (lane < 16) ? 0 : 8;          // A: low lanes K 0..7(+16..23), high +8
    const int kB = (lane < 16) ? 0 : 8;          // B: low lanes K 0..7(+16..23), high +8

    v8f acc = {};
    for (int kc = 0; kc < K; kc += 32) {
        const _Float16* pa = A  + (size_t)rowA * K + kc + kA;
        const _Float16* pb = Bt + (size_t)colB * K + kc + kB;
        v8h a_lo = *reinterpret_cast<const v8h*>(pa);
        v8h a_hi = *reinterpret_cast<const v8h*>(pa + 16);
        v8h b_lo = *reinterpret_cast<const v8h*>(pb);
        v8h b_hi = *reinterpret_cast<const v8h*>(pb + 16);
        v16h a, b;
#pragma unroll
        for (int i = 0; i < 8; ++i) {
            a[i] = a_lo[i]; a[i + 8] = a_hi[i];
            b[i] = b_lo[i]; b[i + 8] = b_hi[i];
        }
        acc = __builtin_amdgcn_wmma_f32_16x16x32_f16(
            /*neg_a=*/false, a, /*neg_b=*/false, b,
            /*c_mod=*/(short)0, acc, /*reuse_a=*/false, /*reuse_b=*/false);
    }

    const int row0 = m0 + ((lane >> 4) << 3);    // lanes 16-31 hold rows m0+8..15
    const int col  = n0 + (lane & 15);
#pragma unroll
    for (int r = 0; r < 8; ++r) {
        int row = row0 + r;
        if (row < M) C[(size_t)row * Ncol + col] = acc[r];
    }
}

// ---------------------------------------------------------------------------
// Aggregation, float4-vectorized (F is 128 or 64; quads per row Q=F/4).
// init: Z[i,f] = bias[f] + dinv[i]^2 * H[i,f]      (self loop + bias)
// edges: Z[dst,f] += dinv[src]*dinv[dst]*H[src,f]  (f32 atomics; Z lives in L2)
// ---------------------------------------------------------------------------
__global__ void k_agg_init4(v4f* __restrict__ Z, const v4f* __restrict__ H,
                            const float* __restrict__ dinv,
                            const v4f* __restrict__ bias4,
                            int N, int logQ) {
    int idx = blockIdx.x * blockDim.x + threadIdx.x;
    int i = idx >> logQ;
    int q = idx & ((1 << logQ) - 1);
    if (i >= N) return;
    float di = dinv[i];
    float s = di * di;
    v4f h = H[idx];
    v4f b = bias4[q];
    v4f z;
    z.x = b.x + s * h.x; z.y = b.y + s * h.y;
    z.z = b.z + s * h.z; z.w = b.w + s * h.w;
    Z[idx] = z;
}

__global__ void k_agg_edges4(float* __restrict__ Z, const v4f* __restrict__ H,
                             const float* __restrict__ dinv,
                             const int* __restrict__ src, const int* __restrict__ dst,
                             int E, int logQ) {
    int idx = blockIdx.x * blockDim.x + threadIdx.x;
    int e = idx >> logQ;
    int q = idx & ((1 << logQ) - 1);
    if (e >= E) return;
    int s = src[e], d = dst[e];
    float nrm = dinv[s] * dinv[d];
    v4f h = H[(((size_t)s) << logQ) + q];        // one b128 load for 4 features
    float* zp = Z + ((((size_t)d) << logQ) + q) * 4;
    atomicAdd(zp + 0, nrm * h.x);
    atomicAdd(zp + 1, nrm * h.y);
    atomicAdd(zp + 2, nrm * h.z);
    atomicAdd(zp + 3, nrm * h.w);
}

// ---------------------------------------------------------------------------
// Host-side orchestration
// ---------------------------------------------------------------------------
static inline size_t align256(size_t v) { return (v + 255) & ~(size_t)255; }

extern "C" void kernel_launch(void* const* d_in, const int* in_sizes, int n_in,
                              void* d_out, int out_size, void* d_ws, size_t ws_size,
                              hipStream_t stream) {
    const float* x  = (const float*)d_in[0];
    const int*   ei = (const int*)d_in[1];     // [2][E] int32 (JAX x64 disabled)
    const float* W1 = (const float*)d_in[2];   // [128][128]
    const float* b1 = (const float*)d_in[3];   // [128]
    const float* W2 = (const float*)d_in[4];   // [128][64]
    const float* b2 = (const float*)d_in[5];   // [64]
    float* out = (float*)d_out;                // [N][64]

    const int INC = 128, HID = 128, OUTC = 64;
    const int N = in_sizes[0] / INC;           // 50000
    const int E = in_sizes[1] / 2;             // 800000
    const int* src = ei;
    const int* dst = ei + E;

    // Workspace carving (with reuse): ~64.3 MB
    char* ws = (char*)d_ws;
    float*    dinv = (float*)ws;      ws += align256((size_t)N * 4);
    _Float16* XH   = (_Float16*)ws;   ws += align256((size_t)N * INC * 2);   // reused as A1h
    _Float16* WT1  = (_Float16*)ws;   ws += align256((size_t)INC * HID * 2);
    _Float16* WT2  = (_Float16*)ws;   ws += align256((size_t)HID * OUTC * 2);
    float*    H    = (float*)ws;      ws += align256((size_t)N * HID * 4);   // H1 then H2
    float*    Z1   = (float*)ws;      ws += align256((size_t)N * HID * 4);

    const int B = 256;
    // 1. degree (self loop counts 1), then dinv = rsqrt(deg)
    k_fill_f32<<<(N + B - 1) / B, B, 0, stream>>>(dinv, 1.0f, N);
    k_deg_accum<<<(E + B - 1) / B, B, 0, stream>>>(dinv, dst, E);
    k_rsqrt_inplace<<<(N + B - 1) / B, B, 0, stream>>>(dinv, N);

    // 2. precision prep
    k_cvt_f32_to_f16<<<((N * INC) + B - 1) / B, B, 0, stream>>>(XH, x, N * INC);
    k_wtrans_f16<<<((INC * HID) + B - 1) / B, B, 0, stream>>>(WT1, W1, INC, HID);
    k_wtrans_f16<<<((HID * OUTC) + B - 1) / B, B, 0, stream>>>(WT2, W2, HID, OUTC);

    const int mtiles = (N + 15) / 16;

    // 3. layer 1: H1 = XH @ W1 (8 waves/block), aggregate (Q=32, logQ=5)
    k_gemm_f16_wmma<<<mtiles, 32 * (HID / 16), 0, stream>>>(XH, WT1, H, N, INC, HID);
    k_agg_init4<<<((N * (HID / 4)) + B - 1) / B, B, 0, stream>>>(
        (v4f*)Z1, (const v4f*)H, dinv, (const v4f*)b1, N, 5);
    k_agg_edges4<<<((E * (HID / 4)) + B - 1) / B, B, 0, stream>>>(
        Z1, (const v4f*)H, dinv, src, dst, E, 5);

    // 4. ReLU -> f16 activations (reuse XH buffer)
    k_relu_cvt_f16<<<((N * HID) + B - 1) / B, B, 0, stream>>>(XH, Z1, N * HID);

    // 5. layer 2: H2 = A1h @ W2 (reuse H buffer), aggregate into d_out (Q=16, logQ=4)
    k_gemm_f16_wmma<<<mtiles, 32 * (OUTC / 16), 0, stream>>>(XH, WT2, H, N, HID, OUTC);
    k_agg_init4<<<((N * (OUTC / 4)) + B - 1) / B, B, 0, stream>>>(
        (v4f*)out, (const v4f*)H, dinv, (const v4f*)b2, N, 4);
    k_agg_edges4<<<((E * (OUTC / 4)) + B - 1) / B, B, 0, stream>>>(
        out, (const v4f*)H, dinv, src, dst, E, 4);
}
